// BaseGNN_59974923321981
// MI455X (gfx1250) — matrix operand order repaired and checked by reading, not compiled, not run on hardware
//
#include <hip/hip_runtime.h>
#include <hip/hip_bf16.h>

// ---------------------------------------------------------------------------
// Problem constants (match reference)
// ---------------------------------------------------------------------------
#define N_NODES 50000
#define N_EDGES 800000
#define F_IN    128
#define DMODEL  256
#define NLAYER  3
#define NGRAPH  512
#define FC_HID  128           // D/2
#define BN_EPS  1e-5f

typedef __attribute__((ext_vector_type(2)))  float  v2f;
typedef __attribute__((ext_vector_type(8)))  float  v8f;
typedef __attribute__((ext_vector_type(16))) __bf16 v16bf;

#if defined(__has_builtin)
#if __has_builtin(__builtin_amdgcn_wmma_f32_16x16x4_f32)
#define USE_F32_WMMA 1
#else
#define USE_F32_WMMA 0
#endif
#else
#define USE_F32_WMMA 0
#endif

// ---------------------------------------------------------------------------
// Tiled WMMA GEMM:  C[M,N] = act( (A (+ A2)) @ B + bias )
//   block = 128 threads (4 waves), tile = 64x64, BK = 32
//   wave w computes rows [w*16, w*16+16) of the tile across all 64 cols
//   Requires: N % 64 == 0, K % 32 == 0.  M handled with bounds checks.
// ---------------------------------------------------------------------------
__global__ __launch_bounds__(128) void gemm_bias_act_kernel(
    const float* __restrict__ A, const float* __restrict__ A2,
    const float* __restrict__ B, const float* __restrict__ bias,
    float* __restrict__ C, int M, int N, int K, int relu) {

  const int mBase = blockIdx.y * 64;
  const int nBase = blockIdx.x * 64;
  const int tid   = threadIdx.x;
  const int lane  = tid & 31;
  const int wave  = tid >> 5;

#if USE_F32_WMMA
  __shared__ float As[64][33];   // 64 rows x 32 k (f32)
  __shared__ float Bs[32][65];   // 32 k x 64 cols (f32)
#else
  __shared__ __bf16 As[64][34];
  __shared__ __bf16 Bs[32][66];
#endif

  v8f acc[4] = {};               // four 16x16 f32 C fragments (this wave's row strip)

  const int mrow = wave * 16 + (lane & 15);   // row within tile for A fragment

  for (int kb = 0; kb < K; kb += 32) {
    // ---- stage A (64x32) with optional elementwise add of A2 ----
    for (int i = tid; i < 64 * 32; i += 128) {
      int r = i >> 5, cc = i & 31;
      int grow = mBase + r;
      float v = 0.f;
      if (grow < M) {
        size_t off = (size_t)grow * K + kb + cc;
        v = A[off];
        if (A2) v += A2[off];
      }
#if USE_F32_WMMA
      As[r][cc] = v;
#else
      As[r][cc] = (__bf16)v;
#endif
    }
    // ---- stage B (32x64) ----
    for (int i = tid; i < 32 * 64; i += 128) {
      int r = i >> 6, cc = i & 63;
      float v = B[(size_t)(kb + r) * N + nBase + cc];
#if USE_F32_WMMA
      Bs[r][cc] = v;
#else
      Bs[r][cc] = (__bf16)v;
#endif
    }
    __syncthreads();

#if USE_F32_WMMA
    // f32 path: K consumed 4 at a time (V_WMMA_F32_16X16X4_F32)
    const int koff = (lane < 16) ? 0 : 2;     // lanes 16-31 hold K+2,K+3
#pragma unroll
    for (int kk = 0; kk < 32; kk += 4) {
      v2f a;
      a.x = As[mrow][kk + koff];
      a.y = As[mrow][kk + koff + 1];
#pragma unroll
      for (int nt = 0; nt < 4; ++nt) {
        int col = nt * 16 + (lane & 15);
        v2f b;
        b.x = Bs[kk + koff][col];
        b.y = Bs[kk + koff + 1][col];
        acc[nt] = __builtin_amdgcn_wmma_f32_16x16x4_f32(
            false, a, false, b, (short)0, acc[nt], false, false);
      }
    }
#else
    // bf16 path: K consumed 32 at a time (V_WMMA_F32_16X16X32_BF16)
    {
      const int k0 = (lane < 16) ? 0 : 8;     // A: lanes 16-31 hold K+8..15 / K+24..31
      v16bf a;
#pragma unroll
      for (int i = 0; i < 8; ++i) {
        a[i]     = As[mrow][k0 + i];
        a[8 + i] = As[mrow][16 + k0 + i];
      }
      const int kb0 = (lane < 16) ? 0 : 8;    // B: lanes 16-31 hold K+8..15 / K+24..31
#pragma unroll
      for (int nt = 0; nt < 4; ++nt) {
        int col = nt * 16 + (lane & 15);
        v16bf b;
#pragma unroll
        for (int h = 0; h < 8; ++h) { b[h] = Bs[kb0 + h][col]; b[8 + h] = Bs[16 + kb0 + h][col]; }
        acc[nt] = __builtin_amdgcn_wmma_f32_16x16x32_bf16(
            false, a, false, b, (short)0, acc[nt], false, false);
      }
    }
#endif
    __syncthreads();
  }

  // ---- epilogue: bias + optional relu, C/D layout per ISA 7.12.2 ----
  const int rbase = mBase + wave * 16 + ((lane < 16) ? 0 : 8);
  const int cbase = nBase + (lane & 15);
#pragma unroll
  for (int nt = 0; nt < 4; ++nt) {
    int col = cbase + nt * 16;
    float bv = bias ? bias[col] : 0.f;
#pragma unroll
    for (int r = 0; r < 8; ++r) {
      int row = rbase + r;
      if (row < M) {
        float v = acc[nt][r] + bv;
        if (relu) v = fmaxf(v, 0.f);
        C[(size_t)row * N + col] = v;
      }
    }
  }
}

// ---------------------------------------------------------------------------
// Helper kernels
// ---------------------------------------------------------------------------
__global__ void fill_zero_kernel(float* __restrict__ p, long n) {
  long i  = (long)blockIdx.x * blockDim.x + threadIdx.x;
  long st = (long)gridDim.x * blockDim.x;
  for (; i < n; i += st) p[i] = 0.f;
}

__global__ void count_nodes_kernel(const int* __restrict__ batch,
                                   float* __restrict__ cnt, int n) {
  int i = blockIdx.x * blockDim.x + threadIdx.x;
  if (i < n) atomicAdd(&cnt[batch[i]], 1.0f);
}

// one wave (32 lanes) per edge: agg[dst] += h[src]
__global__ void scatter_add_kernel(const float* __restrict__ h,
                                   const int* __restrict__ src,
                                   const int* __restrict__ dst,
                                   float* __restrict__ agg, int nedges) {
  int w    = (int)((blockIdx.x * (unsigned)blockDim.x + threadIdx.x) >> 5);
  int lane = threadIdx.x & 31;
  if (w >= nedges) return;
  int s = src[w], d = dst[w];
  const float* hr = h + (size_t)s * DMODEL;
  float*       ar = agg + (size_t)d * DMODEL;
#pragma unroll
  for (int c = lane; c < DMODEL; c += 32) atomicAdd(&ar[c], hr[c]);
}

// column sums / sums of squares over N rows (DMODEL columns)
__global__ void bn_stats_kernel(const float* __restrict__ z,
                                float* __restrict__ sums,
                                float* __restrict__ sumsq, int n) {
  int c = threadIdx.x;  // 256 threads == DMODEL columns
  float s = 0.f, s2 = 0.f;
  for (int r = blockIdx.x; r < n; r += gridDim.x) {
    float v = z[(size_t)r * DMODEL + c];
    s += v;
    s2 += v * v;
  }
  atomicAdd(&sums[c], s);
  atomicAdd(&sumsq[c], s2);
}

// normalize + relu -> h ; accumulate graph-pool sums (block b == row b)
__global__ void bn_apply_pool_kernel(const float* __restrict__ z,
                                     const float* __restrict__ sums,
                                     const float* __restrict__ sumsq,
                                     const float* __restrict__ gamma,
                                     const float* __restrict__ beta,
                                     const int* __restrict__ batch,
                                     float* __restrict__ h,
                                     float* __restrict__ gpool,
                                     int n, int layer) {
  int r = blockIdx.x;
  int c = threadIdx.x;
  if (r >= n) return;
  float invn = 1.0f / (float)n;
  float mean = sums[c] * invn;
  float var  = sumsq[c] * invn - mean * mean;
  size_t idx = (size_t)r * DMODEL + c;
  float v = gamma[c] * (z[idx] - mean) * rsqrtf(var + BN_EPS) + beta[c];
  v = fmaxf(v, 0.f);
  h[idx] = v;
  atomicAdd(&gpool[(size_t)batch[r] * (NLAYER * DMODEL) + layer * DMODEL + c], v);
}

__global__ void pool_div_kernel(float* __restrict__ gpool,
                                const float* __restrict__ cnt, int total) {
  int i = blockIdx.x * blockDim.x + threadIdx.x;
  if (i < total) {
    int g = i / (NLAYER * DMODEL);
    gpool[i] /= fmaxf(cnt[g], 1.0f);
  }
}

// out[g] = dot(hfc[g,:], W_fc2) + b_fc2 ; one wave per graph
__global__ void fc2_kernel(const float* __restrict__ hfc,
                           const float* __restrict__ W,
                           const float* __restrict__ b,
                           float* __restrict__ out, int ngraphs) {
  int g    = (int)((blockIdx.x * (unsigned)blockDim.x + threadIdx.x) >> 5);
  int lane = threadIdx.x & 31;
  if (g >= ngraphs) return;
  float s = 0.f;
  for (int j = lane; j < FC_HID; j += 32) s += hfc[(size_t)g * FC_HID + j] * W[j];
#pragma unroll
  for (int off = 16; off > 0; off >>= 1) s += __shfl_down(s, off, 32);
  if (lane == 0) out[g] = s + b[0];
}

// ---------------------------------------------------------------------------
// Launch
// ---------------------------------------------------------------------------
extern "C" void kernel_launch(void* const* d_in, const int* in_sizes, int n_in,
                              void* d_out, int out_size, void* d_ws, size_t ws_size,
                              hipStream_t stream) {
  (void)in_sizes; (void)n_in; (void)out_size; (void)ws_size;

  const float* x     = (const float*)d_in[0];
  const int*   eidx  = (const int*)d_in[1];
  const int*   batch = (const int*)d_in[2];
  // d_in[3] = num_graphs (compile-time NGRAPH)
  const float* W_enc = (const float*)d_in[4];
  const float* b_enc = (const float*)d_in[5];
  const float* W1    = (const float*)d_in[6];
  const float* b1    = (const float*)d_in[7];
  const float* W2    = (const float*)d_in[8];
  const float* b2    = (const float*)d_in[9];
  const float* gamma = (const float*)d_in[10];
  const float* beta  = (const float*)d_in[11];
  const float* Wfc1  = (const float*)d_in[12];
  const float* bfc1  = (const float*)d_in[13];
  const float* Wfc2  = (const float*)d_in[14];
  const float* bfc2  = (const float*)d_in[15];
  float* out = (float*)d_out;

  // workspace layout (floats)
  float* ws    = (float*)d_ws;
  const size_t ND = (size_t)N_NODES * DMODEL;
  float* h     = ws;                 // N*D   (node features; rewritten each layer)
  float* agg   = h + ND;             // N*D   (neighbor sums; reused as z2)
  float* z1    = agg + ND;           // N*D
  float* gpool = z1 + ND;            // G * L*D
  float* cnt   = gpool + (size_t)NGRAPH * NLAYER * DMODEL;  // G
  float* sums  = cnt + NGRAPH;       // D
  float* sumsq = sums + DMODEL;      // D
  float* hfc   = sumsq + DMODEL;     // G * FC_HID

  // zero pooled sums + counts (contiguous)
  fill_zero_kernel<<<1024, 256, 0, stream>>>(
      gpool, (long)NGRAPH * NLAYER * DMODEL + NGRAPH + 2 * DMODEL);
  count_nodes_kernel<<<(N_NODES + 255) / 256, 256, 0, stream>>>(batch, cnt, N_NODES);

  // encoder: h = x @ W_enc + b_enc
  {
    dim3 grid(DMODEL / 64, (N_NODES + 63) / 64);
    gemm_bias_act_kernel<<<grid, 128, 0, stream>>>(x, nullptr, W_enc, b_enc, h,
                                                   N_NODES, DMODEL, F_IN, 0);
  }

  for (int l = 0; l < NLAYER; ++l) {
    fill_zero_kernel<<<2048, 256, 0, stream>>>(agg, (long)ND);
    // agg[dst] += h[src]
    scatter_add_kernel<<<N_EDGES / 8, 256, 0, stream>>>(h, eidx, eidx + N_EDGES,
                                                        agg, N_EDGES);
    dim3 grid(DMODEL / 64, (N_NODES + 63) / 64);
    // z1 = relu((h + agg) @ W1[l] + b1[l])
    gemm_bias_act_kernel<<<grid, 128, 0, stream>>>(
        h, agg, W1 + (size_t)l * DMODEL * DMODEL, b1 + l * DMODEL, z1,
        N_NODES, DMODEL, DMODEL, 1);
    // z2 (-> agg) = z1 @ W2[l] + b2[l]
    gemm_bias_act_kernel<<<grid, 128, 0, stream>>>(
        z1, nullptr, W2 + (size_t)l * DMODEL * DMODEL, b2 + l * DMODEL, agg,
        N_NODES, DMODEL, DMODEL, 0);
    // batchnorm stats
    fill_zero_kernel<<<2, 256, 0, stream>>>(sums, 2 * DMODEL);
    bn_stats_kernel<<<512, DMODEL, 0, stream>>>(agg, sums, sumsq, N_NODES);
    // normalize + relu -> h, accumulate graph pool
    bn_apply_pool_kernel<<<N_NODES, DMODEL, 0, stream>>>(
        agg, sums, sumsq, gamma + l * DMODEL, beta + l * DMODEL, batch, h, gpool,
        N_NODES, l);
  }

  // pooled means
  pool_div_kernel<<<(NGRAPH * NLAYER * DMODEL + 255) / 256, 256, 0, stream>>>(
      gpool, cnt, NGRAPH * NLAYER * DMODEL);

  // hfc = relu(gpool @ W_fc1 + b_fc1)   [512 x 768] @ [768 x 128]
  {
    dim3 grid(FC_HID / 64, (NGRAPH + 63) / 64);
    gemm_bias_act_kernel<<<grid, 128, 0, stream>>>(gpool, nullptr, Wfc1, bfc1, hfc,
                                                   NGRAPH, FC_HID, NLAYER * DMODEL, 1);
  }
  // out = hfc @ W_fc2 + b_fc2
  fc2_kernel<<<(NGRAPH * 32 + 255) / 256, 256, 0, stream>>>(hfc, Wfc2, bfc2, out,
                                                            NGRAPH);
}
